// TransformerBlock_65352222376159
// MI455X (gfx1250) — compile-verified, matrix operand order checked by reading, not run on hardware
//
#include <hip/hip_runtime.h>

// ---------------------------------------------------------------------------
// Problem constants (match the reference)
// ---------------------------------------------------------------------------
#define NNODES 32768
#define IN_DIM 512
#define HID    512
#define NHEAD  4
#define FFDIM  1024

typedef __bf16 bf16_t;
typedef bf16_t v16bf __attribute__((ext_vector_type(16)));
typedef float  v8f   __attribute__((ext_vector_type(8)));

#define WMMA_BF16(a, b, c) \
    __builtin_amdgcn_wmma_f32_16x16x32_bf16(false, (a), false, (b), (short)0, (c), false, false)

static __device__ __forceinline__ v8f v8zero() {
    v8f z;
#pragma unroll
    for (int i = 0; i < 8; ++i) z[i] = 0.0f;
    return z;
}

// A operand: 16x32 bf16 tile from a row-major matrix.
// Lane L (0..15)  holds row M=L,    K = 0..7  and 16..23
// Lane L (16..31) holds row M=L-16, K = 8..15 and 24..31
static __device__ __forceinline__ v16bf load_a16x32(const bf16_t* __restrict__ A, int lda,
                                                    int rowBase, int kb, int lane) {
    int row  = lane & 15;
    int koff = (lane >> 4) << 3;
    const bf16_t* p = A + (size_t)(rowBase + row) * lda + (kb << 5) + koff;
    union { v16bf v; uint4 q[2]; } u;
    u.q[0] = *reinterpret_cast<const uint4*>(p);
    u.q[1] = *reinterpret_cast<const uint4*>(p + 16);
    return u.v;
}

// B operand: pre-packed 32x16 tile, lane-major (16 bf16 = 32 contiguous bytes/lane)
static __device__ __forceinline__ v16bf load_b_tile(const bf16_t* __restrict__ Bt, int lane) {
    const bf16_t* p = Bt + (lane << 4);
    union { v16bf v; uint4 q[2]; } u;
    u.q[0] = *reinterpret_cast<const uint4*>(p);
    u.q[1] = *reinterpret_cast<const uint4*>(p + 8);
    return u.v;
}

// ---------------------------------------------------------------------------
// 1) f32 -> bf16 conversion (4 elems / thread)
// ---------------------------------------------------------------------------
__global__ void cvt_f32_bf16_kernel(const float* __restrict__ in, bf16_t* __restrict__ out, int n4) {
    int i = blockIdx.x * blockDim.x + threadIdx.x;
    if (i >= n4) return;
    float4 f = reinterpret_cast<const float4*>(in)[i];
    union { bf16_t h[4]; uint2 u; } o;
    o.h[0] = (bf16_t)f.x; o.h[1] = (bf16_t)f.y; o.h[2] = (bf16_t)f.z; o.h[3] = (bf16_t)f.w;
    reinterpret_cast<uint2*>(out)[i] = o.u;
}

// ---------------------------------------------------------------------------
// 2) Pack a row-major [K x Nc] f32 weight into WMMA B-operand tiles (bf16).
//    Tile (nb, kb) stored at out[(nb*KT + kb)*512], lane-major.
// ---------------------------------------------------------------------------
__global__ void pack_b_kernel(const float* __restrict__ W, bf16_t* __restrict__ out,
                              int K, int Nc) {
    int t    = blockIdx.x * blockDim.x + threadIdx.x;
    int tile = t >> 5;
    int lane = t & 31;
    int KT = K >> 5, NT = Nc >> 4;
    if (tile >= KT * NT) return;
    int nb = tile / KT;
    int kb = tile - nb * KT;
    int n     = (nb << 4) + (lane & 15);
    int khalf = (lane >> 4) << 3;
    bf16_t* o = out + ((size_t)tile << 9) + (lane << 4);
    const float* w0 = W + (size_t)((kb << 5) + khalf) * Nc + n;
    const float* w1 = W + (size_t)((kb << 5) + 16 + khalf) * Nc + n;
#pragma unroll
    for (int j = 0; j < 8; ++j) o[j] = (bf16_t)w0[(size_t)j * Nc];
#pragma unroll
    for (int j = 0; j < 8; ++j) o[8 + j] = (bf16_t)w1[(size_t)j * Nc];
}

// ---------------------------------------------------------------------------
// 3) Fused QKV projections + att = (q . k) / sqrt(HID).
//    grid(x = N/16, y = head), block 256 (8 waves). Wave w: cols [64w, 64w+64).
//    Software-pipelined: B tiles of iteration kb grouped ahead of the WMMA
//    burst; A tiles for kb+1 prefetched during the burst.
// ---------------------------------------------------------------------------
__global__ __launch_bounds__(256) void qkv_att_kernel(
    const bf16_t* __restrict__ xb, const bf16_t* __restrict__ hb,
    const bf16_t* __restrict__ wqP, const bf16_t* __restrict__ wkP,
    const bf16_t* __restrict__ wvP,
    bf16_t* __restrict__ vout, float* __restrict__ att) {
    const int lane = threadIdx.x & 31, wave = threadIdx.x >> 5;
    const int rowBase = blockIdx.x << 4;
    const int head = blockIdx.y;
    __shared__ float att_p[8][16];

    const size_t wsz = (size_t)HID * HID;
    const bf16_t* wq = wqP + head * wsz;
    const bf16_t* wk = wkP + head * wsz;
    const bf16_t* wv = wvP + head * wsz;

    v8f cq[4], ck[4], cv[4];
#pragma unroll
    for (int nt = 0; nt < 4; ++nt) { cq[nt] = v8zero(); ck[nt] = v8zero(); cv[nt] = v8zero(); }

    v16bf ax = load_a16x32(xb, IN_DIM, rowBase, 0, lane);
    v16bf ah = load_a16x32(hb, IN_DIM, rowBase, 0, lane);

    for (int kb = 0; kb < IN_DIM / 32; ++kb) {
        v16bf bq[4], bk[4], bv[4];
#pragma unroll
        for (int nt = 0; nt < 4; ++nt) {
            size_t tIdx = (size_t)(((wave << 2) + nt) * (IN_DIM / 32) + kb) << 9;
            bq[nt] = load_b_tile(wq + tIdx, lane);
            bk[nt] = load_b_tile(wk + tIdx, lane);
            bv[nt] = load_b_tile(wv + tIdx, lane);
        }
        int kn = (kb < IN_DIM / 32 - 1) ? kb + 1 : kb;     // clamped, uniform
        v16bf axn = load_a16x32(xb, IN_DIM, rowBase, kn, lane);
        v16bf ahn = load_a16x32(hb, IN_DIM, rowBase, kn, lane);
#pragma unroll
        for (int nt = 0; nt < 4; ++nt) {
            cq[nt] = WMMA_BF16(ax, bq[nt], cq[nt]);
            ck[nt] = WMMA_BF16(ah, bk[nt], ck[nt]);
            cv[nt] = WMMA_BF16(ax, bv[nt], cv[nt]);
        }
        ax = axn; ah = ahn;
    }

    const int hi = lane >> 4, nlo = lane & 15;
    // att partial: reduce q*k over the 16 columns held by this half-wave
#pragma unroll
    for (int r = 0; r < 8; ++r) {
        float s = 0.f;
#pragma unroll
        for (int nt = 0; nt < 4; ++nt) s += cq[nt][r] * ck[nt][r];
        s += __shfl_xor(s, 1); s += __shfl_xor(s, 2);
        s += __shfl_xor(s, 4); s += __shfl_xor(s, 8);
        if (nlo == 0) att_p[wave][(hi << 3) + r] = s;
    }
    // store v (bf16, head-major)
#pragma unroll
    for (int nt = 0; nt < 4; ++nt) {
        int col = (wave << 6) + (nt << 4) + nlo;
#pragma unroll
        for (int r = 0; r < 8; ++r) {
            int row = rowBase + (hi << 3) + r;
            vout[((size_t)head * NNODES + row) * HID + col] = (bf16_t)cv[nt][r];
        }
    }
    __syncthreads();
    if (threadIdx.x < 16) {
        float t = 0.f;
#pragma unroll
        for (int w = 0; w < 8; ++w) t += att_p[w][threadIdx.x];
        att[(size_t)(rowBase + threadIdx.x) * NHEAD + head] = t * 0.04419417382415922f; // 1/sqrt(512)
    }
}

// ---------------------------------------------------------------------------
// 4) Softmax over the node axis (per head). grid = NHEAD, block = 1024.
// ---------------------------------------------------------------------------
__global__ __launch_bounds__(1024) void softmax_nodes_kernel(const float* __restrict__ att,
                                                             float* __restrict__ alpha) {
    const int h = blockIdx.x, t = threadIdx.x;
    __shared__ float red[1024];
    float m = -1e30f;
    for (int n = t; n < NNODES; n += 1024) m = fmaxf(m, att[(size_t)n * NHEAD + h]);
    red[t] = m; __syncthreads();
    for (int s = 512; s > 0; s >>= 1) { if (t < s) red[t] = fmaxf(red[t], red[t + s]); __syncthreads(); }
    m = red[0]; __syncthreads();
    float sum = 0.f;
    for (int n = t; n < NNODES; n += 1024) sum += __expf(att[(size_t)n * NHEAD + h] - m);
    red[t] = sum; __syncthreads();
    for (int s = 512; s > 0; s >>= 1) { if (t < s) red[t] += red[t + s]; __syncthreads(); }
    float inv = 1.f / red[0];
    for (int n = t; n < NNODES; n += 1024)
        alpha[(size_t)n * NHEAD + h] = __expf(att[(size_t)n * NHEAD + h] - m) * inv;
}

// 4b) tp = v * alpha (in place on the bf16 v buffer, head-major), 8 elems/thread
__global__ void scale_v_kernel(bf16_t* __restrict__ v, const float* __restrict__ alpha) {
    size_t i = (size_t)blockIdx.x * blockDim.x + threadIdx.x;   // total/8 threads
    size_t base = i << 3;                                       // 8 contiguous bf16
    int h = (int)(base >> 24);                                  // N*HID = 2^24 per head
    int n = (int)((base >> 9) & (NNODES - 1));
    float a = alpha[(size_t)n * NHEAD + h];
    union { uint4 q; bf16_t e[8]; } u;
    u.q = *reinterpret_cast<const uint4*>(v + base);
#pragma unroll
    for (int j = 0; j < 8; ++j) u.e[j] = (bf16_t)((float)u.e[j] * a);
    *reinterpret_cast<uint4*>(v + base) = u.q;
}

// ---------------------------------------------------------------------------
// 5) mo = tp @ Wc (K = 2048) fused with +x residual and LayerNorm.
//    grid = N/16, block 256; WG owns full 512-wide rows -> LN is block-local.
// ---------------------------------------------------------------------------
__global__ __launch_bounds__(256) void mo_ln_kernel(
    const bf16_t* __restrict__ tp, const bf16_t* __restrict__ wcP,
    const float* __restrict__ x, const float* __restrict__ ln_g, const float* __restrict__ ln_b,
    float* __restrict__ h1f, bf16_t* __restrict__ h1b) {
    const int lane = threadIdx.x & 31, wave = threadIdx.x >> 5;
    const int rowBase = blockIdx.x << 4;
    __shared__ float sum_p[8][16], sq_p[8][16], mu_s[16], rs_s[16];

    v8f acc[4];
#pragma unroll
    for (int nt = 0; nt < 4; ++nt) acc[nt] = v8zero();

    v16bf a = load_a16x32(tp, HID, rowBase, 0, lane);
    for (int kb = 0; kb < 64; ++kb) {            // K = NHEAD*HID = 2048
        v16bf b[4];
#pragma unroll
        for (int nt = 0; nt < 4; ++nt)
            b[nt] = load_b_tile(wcP + ((size_t)(((wave << 2) + nt) * 64 + kb) << 9), lane);
        int kn = (kb < 63) ? kb + 1 : kb;
        v16bf an = load_a16x32(tp + (size_t)(kn >> 4) * NNODES * HID, HID, rowBase, kn & 15, lane);
#pragma unroll
        for (int nt = 0; nt < 4; ++nt) acc[nt] = WMMA_BF16(a, b[nt], acc[nt]);
        a = an;
    }

    const int hi = lane >> 4, nlo = lane & 15;
    // residual x
#pragma unroll
    for (int nt = 0; nt < 4; ++nt) {
        int col = (wave << 6) + (nt << 4) + nlo;
#pragma unroll
        for (int r = 0; r < 8; ++r)
            acc[nt][r] += x[(size_t)(rowBase + (hi << 3) + r) * HID + col];
    }
    // LN statistics (deterministic two-stage reduction)
#pragma unroll
    for (int r = 0; r < 8; ++r) {
        float s = 0.f, s2 = 0.f;
#pragma unroll
        for (int nt = 0; nt < 4; ++nt) { float v = acc[nt][r]; s += v; s2 += v * v; }
        s  += __shfl_xor(s, 1);  s  += __shfl_xor(s, 2);  s  += __shfl_xor(s, 4);  s  += __shfl_xor(s, 8);
        s2 += __shfl_xor(s2, 1); s2 += __shfl_xor(s2, 2); s2 += __shfl_xor(s2, 4); s2 += __shfl_xor(s2, 8);
        if (nlo == 0) { sum_p[wave][(hi << 3) + r] = s; sq_p[wave][(hi << 3) + r] = s2; }
    }
    __syncthreads();
    if (threadIdx.x < 16) {
        float S = 0.f, S2 = 0.f;
#pragma unroll
        for (int w = 0; w < 8; ++w) { S += sum_p[w][threadIdx.x]; S2 += sq_p[w][threadIdx.x]; }
        float mu = S * (1.f / HID);
        mu_s[threadIdx.x] = mu;
        rs_s[threadIdx.x] = rsqrtf(S2 * (1.f / HID) - mu * mu + 1e-5f);
    }
    __syncthreads();
    float mu[8], rs[8];
#pragma unroll
    for (int r = 0; r < 8; ++r) { mu[r] = mu_s[(hi << 3) + r]; rs[r] = rs_s[(hi << 3) + r]; }
#pragma unroll
    for (int nt = 0; nt < 4; ++nt) {
        int col = (wave << 6) + (nt << 4) + nlo;
        float g = ln_g[col], bb = ln_b[col];
#pragma unroll
        for (int r = 0; r < 8; ++r) {
            float val = (acc[nt][r] - mu[r]) * rs[r] * g + bb;
            size_t o = (size_t)(rowBase + (hi << 3) + r) * HID + col;
            h1f[o] = val;
            h1b[o] = (bf16_t)val;
        }
    }
}

// ---------------------------------------------------------------------------
// 6) FF1: relu(h1 @ W1 + b1) -> bf16 [N, 1024]. grid(N/16, 2).
// ---------------------------------------------------------------------------
__global__ __launch_bounds__(256) void ff1_kernel(
    const bf16_t* __restrict__ h1b, const bf16_t* __restrict__ w1P,
    const float* __restrict__ b1, bf16_t* __restrict__ ffb) {
    const int lane = threadIdx.x & 31, wave = threadIdx.x >> 5;
    const int rowBase = blockIdx.x << 4;
    const int colHalf = blockIdx.y;

    v8f acc[4];
#pragma unroll
    for (int nt = 0; nt < 4; ++nt) acc[nt] = v8zero();

    v16bf a = load_a16x32(h1b, HID, rowBase, 0, lane);
    for (int kb = 0; kb < HID / 32; ++kb) {
        v16bf b[4];
#pragma unroll
        for (int nt = 0; nt < 4; ++nt) {
            int ntG = (colHalf << 5) + (wave << 2) + nt;      // 0..63
            b[nt] = load_b_tile(w1P + ((size_t)(ntG * (HID / 32) + kb) << 9), lane);
        }
        int kn = (kb < HID / 32 - 1) ? kb + 1 : kb;
        v16bf an = load_a16x32(h1b, HID, rowBase, kn, lane);
#pragma unroll
        for (int nt = 0; nt < 4; ++nt) acc[nt] = WMMA_BF16(a, b[nt], acc[nt]);
        a = an;
    }
    const int hi = lane >> 4, nlo = lane & 15;
#pragma unroll
    for (int nt = 0; nt < 4; ++nt) {
        int col = (colHalf << 9) + (wave << 6) + (nt << 4) + nlo;
        float bias = b1[col];
#pragma unroll
        for (int r = 0; r < 8; ++r) {
            float v = fmaxf(acc[nt][r] + bias, 0.f);
            ffb[(size_t)(rowBase + (hi << 3) + r) * FFDIM + col] = (bf16_t)v;
        }
    }
}

// ---------------------------------------------------------------------------
// 7) FF2 (K = 1024) + b2 + residual h1 + final LayerNorm -> f32 out.
// ---------------------------------------------------------------------------
__global__ __launch_bounds__(256) void ff2_ln_kernel(
    const bf16_t* __restrict__ ffb, const bf16_t* __restrict__ w2P,
    const float* __restrict__ b2, const float* __restrict__ h1f,
    const float* __restrict__ ln_g, const float* __restrict__ ln_b,
    float* __restrict__ out) {
    const int lane = threadIdx.x & 31, wave = threadIdx.x >> 5;
    const int rowBase = blockIdx.x << 4;
    __shared__ float sum_p[8][16], sq_p[8][16], mu_s[16], rs_s[16];

    v8f acc[4];
#pragma unroll
    for (int nt = 0; nt < 4; ++nt) acc[nt] = v8zero();

    v16bf a = load_a16x32(ffb, FFDIM, rowBase, 0, lane);
    for (int kb = 0; kb < FFDIM / 32; ++kb) {
        v16bf b[4];
#pragma unroll
        for (int nt = 0; nt < 4; ++nt)
            b[nt] = load_b_tile(w2P + ((size_t)(((wave << 2) + nt) * (FFDIM / 32) + kb) << 9), lane);
        int kn = (kb < FFDIM / 32 - 1) ? kb + 1 : kb;
        v16bf an = load_a16x32(ffb, FFDIM, rowBase, kn, lane);
#pragma unroll
        for (int nt = 0; nt < 4; ++nt) acc[nt] = WMMA_BF16(a, b[nt], acc[nt]);
        a = an;
    }
    const int hi = lane >> 4, nlo = lane & 15;
#pragma unroll
    for (int nt = 0; nt < 4; ++nt) {
        int col = (wave << 6) + (nt << 4) + nlo;
        float bias = b2[col];
#pragma unroll
        for (int r = 0; r < 8; ++r)
            acc[nt][r] += bias + h1f[(size_t)(rowBase + (hi << 3) + r) * HID + col];
    }
#pragma unroll
    for (int r = 0; r < 8; ++r) {
        float s = 0.f, s2 = 0.f;
#pragma unroll
        for (int nt = 0; nt < 4; ++nt) { float v = acc[nt][r]; s += v; s2 += v * v; }
        s  += __shfl_xor(s, 1);  s  += __shfl_xor(s, 2);  s  += __shfl_xor(s, 4);  s  += __shfl_xor(s, 8);
        s2 += __shfl_xor(s2, 1); s2 += __shfl_xor(s2, 2); s2 += __shfl_xor(s2, 4); s2 += __shfl_xor(s2, 8);
        if (nlo == 0) { sum_p[wave][(hi << 3) + r] = s; sq_p[wave][(hi << 3) + r] = s2; }
    }
    __syncthreads();
    if (threadIdx.x < 16) {
        float S = 0.f, S2 = 0.f;
#pragma unroll
        for (int w = 0; w < 8; ++w) { S += sum_p[w][threadIdx.x]; S2 += sq_p[w][threadIdx.x]; }
        float mu = S * (1.f / HID);
        mu_s[threadIdx.x] = mu;
        rs_s[threadIdx.x] = rsqrtf(S2 * (1.f / HID) - mu * mu + 1e-5f);
    }
    __syncthreads();
    float mu[8], rs[8];
#pragma unroll
    for (int r = 0; r < 8; ++r) { mu[r] = mu_s[(hi << 3) + r]; rs[r] = rs_s[(hi << 3) + r]; }
#pragma unroll
    for (int nt = 0; nt < 4; ++nt) {
        int col = (wave << 6) + (nt << 4) + nlo;
        float g = ln_g[col], bb = ln_b[col];
#pragma unroll
        for (int r = 0; r < 8; ++r)
            out[(size_t)(rowBase + (hi << 3) + r) * HID + col] =
                (acc[nt][r] - mu[r]) * rs[r] * g + bb;
    }
}

// ---------------------------------------------------------------------------
// Host orchestration
// ---------------------------------------------------------------------------
extern "C" void kernel_launch(void* const* d_in, const int* in_sizes, int n_in,
                              void* d_out, int out_size, void* d_ws, size_t ws_size,
                              hipStream_t stream) {
    const float* x   = (const float*)d_in[0];
    const float* ih  = (const float*)d_in[1];
    // d_in[2]=edge_index, d_in[3]=batch: unused by the reference forward
    const float* Wq  = (const float*)d_in[4];
    const float* Wk  = (const float*)d_in[5];
    const float* Wv  = (const float*)d_in[6];
    const float* Wc  = (const float*)d_in[7];
    const float* W1  = (const float*)d_in[8];
    const float* b1  = (const float*)d_in[9];
    const float* W2  = (const float*)d_in[10];
    const float* b2  = (const float*)d_in[11];
    const float* lng = (const float*)d_in[12];
    const float* lnb = (const float*)d_in[13];
    float* out = (float*)d_out;

    char* p = (char*)d_ws;
    auto alloc = [&](size_t bytes) { void* r = (void*)p; p += (bytes + 255) & ~(size_t)255; return r; };

    const size_t NX = (size_t)NNODES * IN_DIM;                 // 16,777,216
    bf16_t* xb   = (bf16_t*)alloc(NX * 2);                     // 33.5 MB (reused as ffb lo)
    bf16_t* hb   = (bf16_t*)alloc(NX * 2);                     // 33.5 MB (reused as ffb hi)
    bf16_t* wqP  = (bf16_t*)alloc((size_t)NHEAD * HID * HID * 2);
    bf16_t* wkP  = (bf16_t*)alloc((size_t)NHEAD * HID * HID * 2);
    bf16_t* wvP  = (bf16_t*)alloc((size_t)NHEAD * HID * HID * 2);
    bf16_t* wcP  = (bf16_t*)alloc((size_t)NHEAD * HID * HID * 2);
    bf16_t* w1P  = (bf16_t*)alloc((size_t)HID * FFDIM * 2);
    bf16_t* w2P  = (bf16_t*)alloc((size_t)FFDIM * HID * 2);
    bf16_t* vbuf = (bf16_t*)alloc((size_t)NHEAD * NNODES * HID * 2);   // 134 MB
    float*  att  = (float*)alloc((size_t)NNODES * NHEAD * 4);
    float*  alp  = (float*)alloc((size_t)NNODES * NHEAD * 4);
    float*  h1f  = (float*)alloc((size_t)NNODES * HID * 4);            // 67 MB
    bf16_t* h1b  = (bf16_t*)alloc((size_t)NNODES * HID * 2);           // 33.5 MB
    bf16_t* ffb  = xb;   // xb/hb are dead after qkv; their 67 MB holds ff intermediate

    // 1) activations -> bf16
    cvt_f32_bf16_kernel<<<(NX / 4 + 255) / 256, 256, 0, stream>>>(x,  xb, (int)(NX / 4));
    cvt_f32_bf16_kernel<<<(NX / 4 + 255) / 256, 256, 0, stream>>>(ih, hb, (int)(NX / 4));

    // 2) pack weights into WMMA B-tile layout
    for (int h = 0; h < NHEAD; ++h) {
        size_t off = (size_t)h * HID * HID;
        pack_b_kernel<<<64, 256, 0, stream>>>(Wq + off, wqP + off, HID, HID);
        pack_b_kernel<<<64, 256, 0, stream>>>(Wk + off, wkP + off, HID, HID);
        pack_b_kernel<<<64, 256, 0, stream>>>(Wv + off, wvP + off, HID, HID);
    }
    pack_b_kernel<<<256, 256, 0, stream>>>(Wc, wcP, NHEAD * HID, HID);   // 2048 tiles
    pack_b_kernel<<<128, 256, 0, stream>>>(W1, w1P, HID, FFDIM);         // 1024 tiles
    pack_b_kernel<<<128, 256, 0, stream>>>(W2, w2P, FFDIM, HID);         // 1024 tiles

    // 3) fused QKV + att
    qkv_att_kernel<<<dim3(NNODES / 16, NHEAD), 256, 0, stream>>>(xb, hb, wqP, wkP, wvP, vbuf, att);

    // 4) softmax over nodes, then scale v in place (8 elems/thread)
    softmax_nodes_kernel<<<NHEAD, 1024, 0, stream>>>(att, alp);
    {
        size_t total = (size_t)NHEAD * NNODES * HID;   // 2^26
        scale_v_kernel<<<(unsigned)(total / 8 / 256), 256, 0, stream>>>(vbuf, alp);
    }

    // 5) mo GEMM (K=2048) + residual + LN
    mo_ln_kernel<<<NNODES / 16, 256, 0, stream>>>(vbuf, wcP, x, lng, lnb, h1f, h1b);

    // 6) FFN
    ff1_kernel<<<dim3(NNODES / 16, 2), 256, 0, stream>>>(h1b, w1P, b1, ffb);
    ff2_ln_kernel<<<NNODES / 16, 256, 0, stream>>>(ffb, w2P, b2, h1f, lng, lnb, out);

    (void)in_sizes; (void)n_in; (void)out_size; (void)ws_size;
}
